// MemNN_82222853914903
// MI455X (gfx1250) — matrix operand order repaired and checked by reading, not compiled, not run on hardware
//
#include <hip/hip_runtime.h>
#include <hip/hip_bf16.h>

// ---------------------------------------------------------------------------
// MemN2N on MI455X (gfx1250), wave32.
// Re-associated algorithm: match = facts·(Wa u), att = (p·facts)@Wc ->
// ~2.7 GFLOP total; runtime bound by two f32 float4 streaming passes over the
// 128MB facts tensor per hop (L2-resident, 192MB L2). Dense GEMMs run on
// v_wmma_f32_16x16x32_bf16 with uniform-branch fast-path staging (b128 loads).
// ---------------------------------------------------------------------------

typedef __attribute__((ext_vector_type(16))) __bf16 v16bf;
typedef __attribute__((ext_vector_type(8)))  float  v8f;

union Frag { unsigned int w[8]; v16bf v; };

// round-half-up bf16 (cheap: add+shift); tie-break bit dropped intentionally
static __device__ __forceinline__ unsigned int f2bf(float f) {
    union { float f; unsigned int u; } x; x.f = f;
    return (x.u + 0x8000u) >> 16;
}
static __device__ __forceinline__ unsigned int pack2bf(float a, float b) {
    union { float f; unsigned int u; } xa, xb;
    xa.f = a; xb.f = b;
    return ((xa.u + 0x8000u) >> 16) | ((xb.u + 0x8000u) & 0xFFFF0000u);
}

#define LDT 36  // halfs per LDS tile row (even; 72B row stride keeps 8B align)

// C[M x N] = A[M x K] @ B (f32 accumulate, bf16 inputs). M assumed == 64.
// bNmajor==0: B is K x N row-major. bNmajor==1: B is N x K row-major (B^T mul).
// Requires K % 4 == 0, N % 4 == 0 (true for all call sites: 256 / 10000).
__global__ __launch_bounds__(256) void gemm_bf16(
    const float* __restrict__ A, const float* __restrict__ B,
    float* __restrict__ C, const float* __restrict__ bias,
    int M, int N, int K, int bNmajor)
{
    __shared__ __align__(16) unsigned short As[64 * LDT];
    __shared__ __align__(16) unsigned short Bs[64 * LDT];

    const int t    = threadIdx.x;
    const int lane = t & 31;
    const int w    = t >> 5;
    const int mt   = w & 3;          // m-subtile 0..3
    const int ntb  = (w >> 2) << 1;  // first of two n-subtiles
    const int n0   = blockIdx.x * 64;

    v8f c0 = {}; v8f c1 = {};

    const int nchunks = (K + 31) / 32;
    for (int ch = 0; ch < nchunks; ++ch) {
        const int k0 = ch * 32;
        const bool kFull = (k0 + 32 <= K);       // uniform
        const bool nFull = (n0 + 64 <= N);       // uniform

        // ---- stage A chunk (64 x 32) ----
        if (kFull) {
            #pragma unroll
            for (int i = 0; i < 2; ++i) {
                int idx4 = t + i * 256;            // 0..511 float4 tiles
                int r  = idx4 >> 3;
                int kq = (idx4 & 7) << 2;
                float4 v = *(const float4*)(A + (long)r * K + k0 + kq);
                uint2 pk = make_uint2(pack2bf(v.x, v.y), pack2bf(v.z, v.w));
                *(uint2*)&As[r * LDT + kq] = pk;
            }
        } else {
            #pragma unroll
            for (int i = 0; i < 8; ++i) {
                int idx = t + i * 256;
                int r = idx >> 5, k = idx & 31;
                float v = ((k0 + k) < K) ? A[(long)r * K + k0 + k] : 0.f;
                As[r * LDT + k] = (unsigned short)f2bf(v);
            }
        }
        // ---- stage B chunk into Bs[n][k] ----
        if (bNmajor) {
            if (kFull && nFull) {
                #pragma unroll
                for (int i = 0; i < 2; ++i) {
                    int idx4 = t + i * 256;
                    int n  = idx4 >> 3;
                    int kq = (idx4 & 7) << 2;
                    float4 v = *(const float4*)(B + (long)(n0 + n) * K + k0 + kq);
                    uint2 pk = make_uint2(pack2bf(v.x, v.y), pack2bf(v.z, v.w));
                    *(uint2*)&Bs[n * LDT + kq] = pk;
                }
            } else {
                #pragma unroll
                for (int i = 0; i < 8; ++i) {
                    int idx = t + i * 256;
                    int n = idx >> 5, k = idx & 31;
                    float v = ((n0 + n) < N && (k0 + k) < K)
                                  ? B[(long)(n0 + n) * K + k0 + k] : 0.f;
                    Bs[n * LDT + k] = (unsigned short)f2bf(v);
                }
            }
        } else {
            if (kFull && nFull) {
                #pragma unroll
                for (int i = 0; i < 2; ++i) {
                    int idx4 = t + i * 256;
                    int k  = idx4 >> 4;            // 16 float4 per 64-wide n row
                    int nq = (idx4 & 15) << 2;
                    float4 v = *(const float4*)(B + (long)(k0 + k) * N + n0 + nq);
                    Bs[(nq + 0) * LDT + k] = (unsigned short)f2bf(v.x);
                    Bs[(nq + 1) * LDT + k] = (unsigned short)f2bf(v.y);
                    Bs[(nq + 2) * LDT + k] = (unsigned short)f2bf(v.z);
                    Bs[(nq + 3) * LDT + k] = (unsigned short)f2bf(v.w);
                }
            } else {
                #pragma unroll
                for (int i = 0; i < 8; ++i) {
                    int idx = t + i * 256;
                    int k = idx >> 6, n = idx & 63;
                    float v = ((k0 + k) < K && (n0 + n) < N)
                                  ? B[(long)(k0 + k) * N + n0 + n] : 0.f;
                    Bs[n * LDT + k] = (unsigned short)f2bf(v);
                }
            }
        }
        __syncthreads();

        // ---- gather WMMA fragments (contiguous k = 16*(lane/16) + s, same for A and B) ----
        Frag af, bf0, bf1;
        const unsigned short* ap  = &As[(mt * 16) * LDT];
        const unsigned short* bp0 = &Bs[(ntb * 16) * LDT];
        const unsigned short* bp1 = &Bs[((ntb + 1) * 16) * LDT];
        const int row = lane & 15;
        #pragma unroll
        for (int p = 0; p < 8; ++p) {
            int kA = 2 * p + ((lane < 16) ? 0 : 16);
            af.w[p] = *(const unsigned int*)(ap + row * LDT + kA);
            int kB = 2 * p + ((lane < 16) ? 0 : 16);
            bf0.w[p] = *(const unsigned int*)(bp0 + row * LDT + kB);
            bf1.w[p] = *(const unsigned int*)(bp1 + row * LDT + kB);
        }
        c0 = __builtin_amdgcn_wmma_f32_16x16x32_bf16(false, af.v, false, bf0.v,
                                                     (short)0, c0, false, false);
        c1 = __builtin_amdgcn_wmma_f32_16x16x32_bf16(false, af.v, false, bf1.v,
                                                     (short)0, c1, false, false);
        __syncthreads();
    }

    // ---- store (C/D layout: vgpr i -> m = i + 8*(lane>=16), n = lane%16) ----
    const bool hasBias = (bias != nullptr);
    #pragma unroll
    for (int i = 0; i < 8; ++i) {
        int m = mt * 16 + i + ((lane >= 16) ? 8 : 0);
        int gn0 = n0 + ntb * 16 + (lane & 15);
        int gn1 = n0 + (ntb + 1) * 16 + (lane & 15);
        if (m < M) {
            if (gn0 < N) C[(long)m * N + gn0] = c0[i] + (hasBias ? bias[gn0] : 0.f);
            if (gn1 < N) C[(long)m * N + gn1] = c1[i] + (hasBias ? bias[gn1] : 0.f);
        }
    }
}

// match[b,l] = facts[b,l,:] . qa[b,:]  (float4 streaming pass; V % 4 == 0)
__global__ __launch_bounds__(256) void match_dot(
    const float* __restrict__ facts, const float* __restrict__ qa,
    float* __restrict__ match, int L, int V)
{
    const int bl = blockIdx.x;
    const int b = bl / L, l = bl % L;
    const int t = threadIdx.x, lane = t & 31, w = t >> 5;
    const float4* fr = (const float4*)(facts + ((long)b * L + l) * V);
    const float4* qr = (const float4*)(qa + (long)b * V);
    const int V4 = V >> 2;
    float acc = 0.f;
    for (int v = t; v < V4; v += 256) {
        float4 f = fr[v], q = qr[v];
        acc += f.x * q.x + f.y * q.y + f.z * q.z + f.w * q.w;
    }
    #pragma unroll
    for (int s = 16; s > 0; s >>= 1) acc += __shfl_xor(acc, s, 32);
    __shared__ float red[8];
    if (lane == 0) red[w] = acc;
    __syncthreads();
    if (t == 0) {
        float r = 0.f;
        #pragma unroll
        for (int i = 0; i < 8; ++i) r += red[i];
        match[bl] = r;
    }
}

// row-wise softmax, blockDim.x >= ncols and power of two
__global__ void softmax_row(const float* __restrict__ in, float* __restrict__ out, int ncols)
{
    extern __shared__ float sm[];
    const int b = blockIdx.x, t = threadIdx.x;
    float v = (t < ncols) ? in[(long)b * ncols + t] : -3.0e38f;
    sm[t] = v; __syncthreads();
    for (int s = blockDim.x >> 1; s > 0; s >>= 1) {
        if (t < s) sm[t] = fmaxf(sm[t], sm[t + s]);
        __syncthreads();
    }
    float mx = sm[0]; __syncthreads();
    float e = (t < ncols) ? __expf(v - mx) : 0.f;
    sm[t] = e; __syncthreads();
    for (int s = blockDim.x >> 1; s > 0; s >>= 1) {
        if (t < s) sm[t] += sm[t + s];
        __syncthreads();
    }
    if (t < ncols) out[(long)b * ncols + t] = e / sm[0];
}

// fbar[b,v] = sum_l p[b,l] * facts[b,l,v]  (float4 streaming pass)
__global__ __launch_bounds__(256) void fbar_kernel(
    const float* __restrict__ facts, const float* __restrict__ p,
    float* __restrict__ fbar, int L, int V)
{
    const int b = blockIdx.y;
    const int V4 = V >> 2;
    const int v4 = blockIdx.x * 256 + threadIdx.x;
    __shared__ float ps[64];
    if (threadIdx.x < L) ps[threadIdx.x] = p[b * L + threadIdx.x];
    __syncthreads();
    if (v4 < V4) {
        const float4* fb = (const float4*)(facts + (long)b * L * V) + v4;
        float4 acc = make_float4(0.f, 0.f, 0.f, 0.f);
        #pragma unroll 5
        for (int l = 0; l < L; ++l) {
            float4 f = fb[(long)l * V4];
            float pl = ps[l];
            acc.x += f.x * pl; acc.y += f.y * pl;
            acc.z += f.z * pl; acc.w += f.w * pl;
        }
        ((float4*)(fbar + (long)b * V))[v4] = acc;
    }
}

__global__ void addvec4(const float4* a, const float4* b, float4* c, int n4) {
    int i = blockIdx.x * 256 + threadIdx.x;
    if (i < n4) {
        float4 x = a[i], y = b[i];
        c[i] = make_float4(x.x + y.x, x.y + y.y, x.z + y.z, x.w + y.w);
    }
}
__global__ void relu4(const float4* a, float4* c, int n4) {
    int i = blockIdx.x * 256 + threadIdx.x;
    if (i < n4) {
        float4 x = a[i];
        c[i] = make_float4(fmaxf(x.x, 0.f), fmaxf(x.y, 0.f),
                           fmaxf(x.z, 0.f), fmaxf(x.w, 0.f));
    }
}

extern "C" void kernel_launch(void* const* d_in, const int* in_sizes, int n_in,
                              void* d_out, int out_size, void* d_ws, size_t ws_size,
                              hipStream_t stream) {
    (void)in_sizes; (void)n_in; (void)out_size; (void)ws_size;
    const float* facts    = (const float*)d_in[0];  // (64,50,10000)
    const float* question = (const float*)d_in[1];  // (64,1,10000); sum over axis1 == itself
    const float* Wq       = (const float*)d_in[2];  // (10000,256)
    const float* Wa       = (const float*)d_in[3];  // (3,10000,256)
    const float* Wc       = (const float*)d_in[4];  // (3,10000,256)
    const float* Ww       = (const float*)d_in[5];  // (3,256,256)
    const float* bw       = (const float*)d_in[6];  // (3,256)
    float* out = (float*)d_out;                     // (64,256)

    const int B = 64, L = 50, V = 10000, D = 256, HOPS = 3;

    float* ws    = (float*)d_ws;
    float* u     = ws;                    // B*D
    float* x     = u + B * D;             // B*D
    float* z     = x + B * D;             // B*D
    float* att   = z + B * D;             // B*D
    float* qa    = att + B * D;           // B*V
    float* fbar  = qa + (long)B * V;      // B*V
    float* match = fbar + (long)B * V;    // B*L
    float* p     = match + B * L;         // B*L

    const dim3 blk(256);
    const int n4 = (B * D) / 4;           // 4096
    const dim3 g4((n4 + 255) / 256);      // 16 blocks

    // u0 = question @ Wq   (M=64, N=256, K=10000)
    gemm_bf16<<<dim3((D + 63) / 64), blk, 0, stream>>>(question, Wq, u, nullptr, B, D, V, 0);

    for (int h = 0; h < HOPS; ++h) {
        // qa = u @ Wa[h]^T  (M=64, N=10000, K=256)
        gemm_bf16<<<dim3((V + 63) / 64), blk, 0, stream>>>(
            u, Wa + (long)h * V * D, qa, nullptr, B, V, D, 1);
        // match[b,l] = facts . qa
        match_dot<<<dim3(B * L), blk, 0, stream>>>(facts, qa, match, L, V);
        // p = softmax over L
        softmax_row<<<dim3(B), dim3(64), 64 * sizeof(float), stream>>>(match, p, L);
        // fbar[b,v] = sum_l p * facts
        fbar_kernel<<<dim3((V / 4 + 255) / 256, B), blk, 0, stream>>>(facts, p, fbar, L, V);
        // att = fbar @ Wc[h]  (M=64, N=256, K=10000)
        gemm_bf16<<<dim3((D + 63) / 64), blk, 0, stream>>>(
            fbar, Wc + (long)h * V * D, att, nullptr, B, D, V, 0);
        // x = u + att
        addvec4<<<g4, blk, 0, stream>>>((const float4*)u, (const float4*)att,
                                        (float4*)x, n4);
        // z = x @ Ww[h] + bw[h]  (M=64, N=256, K=256)
        gemm_bf16<<<dim3((D + 63) / 64), blk, 0, stream>>>(
            x, Ww + (long)h * D * D, z, bw + (long)h * D, B, D, D, 0);
        if (h < HOPS - 1)
            relu4<<<g4, blk, 0, stream>>>((const float4*)z, (float4*)u, n4);
        else
            softmax_row<<<dim3(B), dim3(256), 256 * sizeof(float), stream>>>(z, out, D);
    }
}
